// Model_23235773071374
// MI455X (gfx1250) — matrix operand order repaired and checked by reading, not compiled, not run on hardware
//
#include <hip/hip_runtime.h>

typedef _Float16 f16;
typedef f16   v16h __attribute__((ext_vector_type(16)));
typedef float v8f  __attribute__((ext_vector_type(8)));

#define DDIM 128          // SIREN hidden width
#define LSTR 136          // padded LDS row stride in halves (128 + 8)
#define NBANDS 16
#define CCH 64
#define EDIM 1024
#define BB 32
#define TIN 512
#define TOUT 256
#define LIN_IN_DIM 1092   // 17*64 + 4
#define LIN_OUT_DIM 1088  // 17*64

// ---------------------------------------------------------------------------
// One 128-row chunk GEMM: Xout = sin(Xin @ W^T + bias), all in LDS (f16),
// f32 WMMA accumulation. 8 waves, wave w owns M-tile w (rows w*16..w*16+15).
// Bias is folded into the accumulator init (C/D layout: one column per lane).
// Activation uses the hardware v_sin_f32 path (__sinf).
// ---------------------------------------------------------------------------
static __device__ __forceinline__ void gemm_sin_128(
    const f16* __restrict__ Xin,   // [128][LSTR] f16 activations
    const f16* __restrict__ Wmat,  // [128][LSTR] f16, Wmat[o][i]
    const float* __restrict__ bias,
    f16* __restrict__ Xout,        // [128][LSTR]
    int wave, int lane)
{
    const int m0 = wave * 16;
    const int hl = lane & 15;      // row-in-tile (A) / column (B, C/D)
    const int hi = lane >> 4;      // lane-half selector

    // A fragments: 16x32 f16, M = m0+hl across lanes 0-15.
    // Per ISA: VGPRs 0-3 hold K = hi*8 + 0..7, VGPRs 4-7 hold K = 16 + hi*8 + 0..7.
    v16h Af[4];
    const f16* arow = Xin + (m0 + hl) * LSTR;
    #pragma unroll
    for (int kk = 0; kk < 4; ++kk) {
        const int kb = kk * 32 + hi * 8;
        v16h a;
        #pragma unroll
        for (int j = 0; j < 8; ++j) a[j]     = arow[kb + j];
        #pragma unroll
        for (int j = 0; j < 8; ++j) a[8 + j] = arow[kb + 16 + j];
        Af[kk] = a;
    }

    #pragma unroll
    for (int nt = 0; nt < 8; ++nt) {
        const int n0 = nt * 16;
        const int o  = n0 + hl;
        const float bval = bias[o];
        // Fold bias into accumulator: lane carries column o for all 8 rows.
        v8f acc = {bval, bval, bval, bval, bval, bval, bval, bval};
        // B fragment: 32x16, N = n0+hl across lanes; B[k][n] = W[n][k].
        // Elements 0-7 carry K = hi*8 + 0..7, elements 8-15 carry K = 16 + hi*8 + 0..7 (per slice).
        const f16* brow = Wmat + (n0 + hl) * LSTR;
        #pragma unroll
        for (int kk = 0; kk < 4; ++kk) {
            const int kb = kk * 32 + hi * 8;
            v16h b;
            #pragma unroll
            for (int j = 0; j < 8; ++j) b[j]     = brow[kb + j];
            #pragma unroll
            for (int j = 0; j < 8; ++j) b[8 + j] = brow[kb + 16 + j];
            acc = __builtin_amdgcn_wmma_f32_16x16x32_f16(
                false, Af[kk], false, b, (short)0, acc, false, false);
        }
        // C/D layout: N = n0+hl, VGPR r -> M = r + hi*8. Epilogue: hw sin.
        #pragma unroll
        for (int r = 0; r < 8; ++r) {
            const int m = r + hi * 8;
            Xout[(m0 + m) * LSTR + o] = (f16)__sinf(acc[r]);
        }
    }
}

// ---------------------------------------------------------------------------
// Per-eigenfunction SIREN evaluation. One block per e; weights f16 in LDS.
// Writes raw final-layer output yraw[e*T + t].
// ---------------------------------------------------------------------------
extern "C" __global__ void __launch_bounds__(256)
siren_eigen_kernel(const float* __restrict__ W0,   // (E, D)
                   const float* __restrict__ b0,   // (E, D)
                   const float* __restrict__ Wh,   // (2, E, D, D)
                   const float* __restrict__ bh,   // (2, E, D)
                   const float* __restrict__ Wl,   // (E, D)
                   const float* __restrict__ bl,   // (E,)
                   const float* __restrict__ w0i,  // (E,)
                   float* __restrict__ yraw,       // (E, T)
                   int T)
{
    extern __shared__ char smem[];
    f16* Wh0s = (f16*)smem;                       // [128][LSTR]
    f16* Wh1s = Wh0s + DDIM * LSTR;
    f16* Xa   = Wh1s + DDIM * LSTR;
    f16* Xb   = Xa   + DDIM * LSTR;
    float* bh0s = (float*)(Xb + DDIM * LSTR);
    float* bh1s = bh0s + DDIM;
    float* wls  = bh1s + DDIM;

    const int e    = blockIdx.x;
    const int tid  = threadIdx.x;
    const int lane = tid & 31;
    const int wave = tid >> 5;

    // Stage both hidden-layer weight matrices (f32 -> f16) once per block.
    const float* Wh0g = Wh + ((size_t)e) * DDIM * DDIM;
    const float* Wh1g = Wh + ((size_t)EDIM + e) * DDIM * DDIM;
    for (int idx = tid; idx < DDIM * DDIM; idx += 256) {
        const int o = idx >> 7, i = idx & 127;
        Wh0s[o * LSTR + i] = (f16)Wh0g[idx];
        Wh1s[o * LSTR + i] = (f16)Wh1g[idx];
    }
    for (int idx = tid; idx < DDIM; idx += 256) {
        bh0s[idx] = bh[((size_t)e) * DDIM + idx];
        bh1s[idx] = bh[((size_t)EDIM + e) * DDIM + idx];
        wls[idx]  = Wl[(size_t)e * DDIM + idx];
    }
    const float w0  = w0i[e];
    const float blv = bl[e];
    const float inv = 1.0f / (float)(T - 1);
    __syncthreads();

    for (int tbase = 0; tbase < T; tbase += 128) {
        // Layer 0: Xa[tl][o] = sin(w0 * (W0[e][o]*dom + b0[e][o]))
        for (int idx = tid; idx < 128 * DDIM; idx += 256) {
            const int tl = idx >> 7, o = idx & 127;
            const float dom = (float)(tbase + tl) * inv;
            const float y = W0[(size_t)e * DDIM + o] * dom + b0[(size_t)e * DDIM + o];
            Xa[tl * LSTR + o] = (f16)__sinf(w0 * y);
        }
        __syncthreads();
        gemm_sin_128(Xa, Wh0s, bh0s, Xb, wave, lane);
        __syncthreads();
        gemm_sin_128(Xb, Wh1s, bh1s, Xa, wave, lane);
        __syncthreads();
        // Final layer: y[t] = dot(Xa[tl], Wl[e]) + bl[e]
        if (tid < 128) {
            float acc = 0.0f;
            const f16* xrow = Xa + tid * LSTR;
            #pragma unroll 8
            for (int i = 0; i < DDIM; ++i) acc += (float)xrow[i] * wls[i];
            yraw[(size_t)e * T + tbase + tid] = acc + blv;
        }
        __syncthreads();
    }
}

// ---------------------------------------------------------------------------
// ef[t][nb][c] with prepended ones row, normalized over the 17 bands.
// ---------------------------------------------------------------------------
extern "C" __global__ void normalize_kernel(const float* __restrict__ yraw, // (E, T)
                                            float* __restrict__ ef,         // (T, 17, C)
                                            int T)
{
    const int gid = blockIdx.x * blockDim.x + threadIdx.x;
    if (gid >= T * CCH) return;
    const int t = gid / CCH, c = gid % CCH;
    float v[NBANDS];
    float s = 1.0f;
    #pragma unroll
    for (int nb = 0; nb < NBANDS; ++nb) {
        const float x = yraw[(size_t)(nb * CCH + c) * T + t];
        v[nb] = x;
        s += x * x;
    }
    const float invn = 1.0f / sqrtf(s);
    ef[((size_t)t * 17 + 0) * CCH + c] = invn;
    #pragma unroll
    for (int nb = 0; nb < NBANDS; ++nb)
        ef[((size_t)t * 17 + nb + 1) * CCH + c] = v[nb] * invn;
}

// coeffs[b,e,c] = sum_t x_enc[b,t,c]*ef_in[t,e,c]; append label -> flat_in (B,1092)
extern "C" __global__ void coeffs_kernel(const float* __restrict__ x_enc,
                                         const float* __restrict__ x_mark_enc,
                                         const float* __restrict__ ef_in,
                                         float* __restrict__ flat_in)
{
    const int gid = blockIdx.x * blockDim.x + threadIdx.x;
    if (gid >= BB * LIN_IN_DIM) return;
    const int b = gid / LIN_IN_DIM, k = gid % LIN_IN_DIM;
    if (k < LIN_OUT_DIM) {
        const int e2 = k / CCH, c = k % CCH;
        float acc = 0.0f;
        for (int t = 0; t < TIN; ++t)
            acc += x_enc[((size_t)b * TIN + t) * CCH + c] *
                   ef_in[((size_t)t * 17 + e2) * CCH + c];
        flat_in[gid] = acc;
    } else {
        flat_in[gid] = x_mark_enc[((size_t)b * TIN + (TIN - 1)) * 4 + (k - LIN_OUT_DIM)];
    }
}

extern "C" __global__ void linear_kernel(const float* __restrict__ flat_in,
                                         const float* __restrict__ Wlin, // (1088,1092)
                                         const float* __restrict__ blin,
                                         float* __restrict__ pred)       // (B,1088)
{
    const int gid = blockIdx.x * blockDim.x + threadIdx.x;
    if (gid >= BB * LIN_OUT_DIM) return;
    const int b = gid / LIN_OUT_DIM, j = gid % LIN_OUT_DIM;
    float acc = blin[j];
    const float* frow = flat_in + (size_t)b * LIN_IN_DIM;
    const float* wrow = Wlin + (size_t)j * LIN_IN_DIM;
    for (int k = 0; k < LIN_IN_DIM; ++k) acc += frow[k] * wrow[k];
    pred[gid] = acc;
}

// out[b,t,c] = sum_e pred[b,e*64+c] * ef_out[t,e,c]
extern "C" __global__ void recon_kernel(const float* __restrict__ pred,
                                        const float* __restrict__ ef_out,
                                        float* __restrict__ out)
{
    const int gid = blockIdx.x * blockDim.x + threadIdx.x;
    if (gid >= BB * TOUT * CCH) return;
    const int c = gid & (CCH - 1);
    const int t = (gid >> 6) & (TOUT - 1);
    const int b = gid >> 14;
    float acc = 0.0f;
    #pragma unroll
    for (int e2 = 0; e2 < 17; ++e2)
        acc += pred[(size_t)b * LIN_OUT_DIM + e2 * CCH + c] *
               ef_out[((size_t)t * 17 + e2) * CCH + c];
    out[gid] = acc;
}

// ---------------------------------------------------------------------------
extern "C" void kernel_launch(void* const* d_in, const int* in_sizes, int n_in,
                              void* d_out, int out_size, void* d_ws, size_t ws_size,
                              hipStream_t stream)
{
    (void)in_sizes; (void)n_in; (void)out_size; (void)ws_size;
    const float* x_enc      = (const float*)d_in[0];
    const float* x_mark_enc = (const float*)d_in[1];
    const float* W0_in  = (const float*)d_in[4];
    const float* b0_in  = (const float*)d_in[5];
    const float* Wh_in  = (const float*)d_in[6];
    const float* bh_in  = (const float*)d_in[7];
    const float* Wl_in  = (const float*)d_in[8];
    const float* bl_in  = (const float*)d_in[9];
    const float* w0i_in = (const float*)d_in[10];
    const float* W0_out  = (const float*)d_in[11];
    const float* b0_out  = (const float*)d_in[12];
    const float* Wh_out  = (const float*)d_in[13];
    const float* bh_out  = (const float*)d_in[14];
    const float* Wl_out  = (const float*)d_in[15];
    const float* bl_out  = (const float*)d_in[16];
    const float* w0i_out = (const float*)d_in[17];
    const float* Wlin = (const float*)d_in[18];
    const float* blin = (const float*)d_in[19];

    float* ws       = (float*)d_ws;
    float* yraw_in  = ws;                                   // 1024*512
    float* yraw_out = yraw_in + (size_t)EDIM * TIN;         // 1024*256
    float* ef_in    = yraw_out + (size_t)EDIM * TOUT;       // 512*17*64
    float* ef_out   = ef_in + (size_t)TIN * 17 * CCH;       // 256*17*64
    float* flat_in  = ef_out + (size_t)TOUT * 17 * CCH;     // 32*1092
    float* pred     = flat_in + (size_t)BB * LIN_IN_DIM;    // 32*1088

    const size_t shbytes = (size_t)4 * DDIM * LSTR * sizeof(_Float16)
                         + (size_t)3 * DDIM * sizeof(float);

    siren_eigen_kernel<<<EDIM, 256, shbytes, stream>>>(
        W0_in, b0_in, Wh_in, bh_in, Wl_in, bl_in, w0i_in, yraw_in, TIN);
    siren_eigen_kernel<<<EDIM, 256, shbytes, stream>>>(
        W0_out, b0_out, Wh_out, bh_out, Wl_out, bl_out, w0i_out, yraw_out, TOUT);

    normalize_kernel<<<(TIN * CCH + 255) / 256, 256, 0, stream>>>(yraw_in, ef_in, TIN);
    normalize_kernel<<<(TOUT * CCH + 255) / 256, 256, 0, stream>>>(yraw_out, ef_out, TOUT);

    coeffs_kernel<<<(BB * LIN_IN_DIM + 255) / 256, 256, 0, stream>>>(
        x_enc, x_mark_enc, ef_in, flat_in);
    linear_kernel<<<(BB * LIN_OUT_DIM + 255) / 256, 256, 0, stream>>>(
        flat_in, Wlin, blin, pred);
    recon_kernel<<<(BB * TOUT * CCH + 255) / 256, 256, 0, stream>>>(
        pred, ef_out, (float*)d_out);
}
